// RPN_12369505813076
// MI455X (gfx1250) — matrix operand order repaired and checked by reading, not compiled, not run on hardware
//
#include <hip/hip_runtime.h>

// ---------- CDNA5 types ----------
typedef __attribute__((ext_vector_type(16))) __bf16 v16bf;
typedef __attribute__((ext_vector_type(8)))  float  v8f;
typedef __attribute__((ext_vector_type(4)))  unsigned int u32x4;
typedef __attribute__((ext_vector_type(4)))  float  f32x4;

union Frag { v16bf v; u32x4 q[2]; };

#define NEGV (-1.0e10f)
#define ROWS 72   // 64 bf16 K-values + 8 pad (bank-conflict-free, 16B-aligned rows)

// fp32 -> bf16 (round to nearest even)
__device__ __forceinline__ unsigned short f2bf(float f) {
  unsigned int u = __float_as_uint(f);
  unsigned int r = u + 0x7fffu + ((u >> 16) & 1u);
  return (unsigned short)(r >> 16);
}

__device__ __forceinline__ v8f wmma_bf16(const Frag& a, const Frag& b, v8f c) {
  return __builtin_amdgcn_wmma_f32_16x16x32_bf16(false, a.v, false, b.v,
                                                 (short)0, c, false, false);
}

// =====================================================================
// Kernel 1: 3x3 conv (implicit GEMM, bf16 WMMA, fp32 acc) + bias + ReLU
// features NCHW [8,512,64,64] fp32 -> featws channels-last bf16 [32768,512]
// Block tile: M=64 (one (b,y) row), N=64 ch.  K stepped 64 (72 stages),
// register double-buffered staging, 4 WMMAs / wave / stage.
// =====================================================================
__global__ __launch_bounds__(256) void k_conv3x3_wmma(
    const float* __restrict__ x, const float* __restrict__ Wc,
    const float* __restrict__ bc, unsigned short* __restrict__ featws) {
  __shared__ __attribute__((aligned(16))) unsigned short Ash[64 * ROWS];
  __shared__ __attribute__((aligned(16))) unsigned short Bsh[64 * ROWS];

  const int tid   = threadIdx.x;
  const int mtile = blockIdx.x;          // 0..511  -> (b, y)
  const int ntile = blockIdx.y;          // 0..7    -> output-channel block
  const int b = mtile >> 6, y = mtile & 63;
  const int lane = tid & 31, wv = tid >> 5;
  const int mw = (wv & 3) << 4;          // wave M sub-tile (0,16,32,48)
  const int nw = (wv >> 2) << 5;         // wave N sub-tile (0,32); covers nw, nw+16

  v8f acc0 = {}; v8f acc1 = {};

  const int ml  = tid & 63;              // A-stage: pixel (== x coordinate)
  const int cs  = tid >> 6;              // A-stage: channel slot (0..3)
  const int nB  = tid & 63;              // B-stage: output channel
  const int csB = tid >> 6;

  float aR[16], bR[16];

  auto loadStage = [&](int kt) {
    const int s  = kt >> 3;              // tap 0..8 (8 K-64 chunks per tap)
    const int c0 = (kt & 7) << 6;        // input-channel block base
    const int dy = s / 3, dx = s - dy * 3;
    const int yy = y + dy - 1;
    const int xx = ml + dx - 1;
    const bool ok = (yy >= 0) & (yy < 64) & (xx >= 0) & (xx < 64);
#pragma unroll
    for (int j = 0; j < 16; ++j) {
      const int ci = cs + (j << 2);
      aR[j] = ok ? x[(((b << 9) + c0 + ci) << 12) + (yy << 6) + xx] : 0.0f;
    }
#pragma unroll
    for (int j = 0; j < 16; ++j) {
      const int ci = (csB << 4) + j;
      bR[j] = Wc[((ntile << 6) + nB) * 4608 + (c0 + ci) * 9 + s];
    }
  };

  loadStage(0);
  for (int kt = 0; kt < 72; ++kt) {
    // commit staged registers to LDS (bf16)
#pragma unroll
    for (int j = 0; j < 16; ++j)
      Ash[ml * ROWS + cs + (j << 2)] = f2bf(aR[j]);
#pragma unroll
    for (int j = 0; j < 16; ++j)
      Bsh[nB * ROWS + (csB << 4) + j] = f2bf(bR[j]);
    __syncthreads();

    // issue next stage's global loads; latency hides behind WMMA + ds_loads
    if (kt + 1 < 72) loadStage(kt + 1);
    // prefetch weights two stages ahead (gfx1250 global_prefetch_b8)
    if (kt + 2 < 72) {
      const int sn = (kt + 2) >> 3, c0n = ((kt + 2) & 7) << 6;
      __builtin_prefetch(&Wc[((ntile << 6) + nB) * 4608 + (c0n + (csB << 4)) * 9 + sn], 0, 1);
    }

    // fragments in contiguous k order, k = 16*(lane/16) + s (A 16x32 bf16, B transposed)
    const int am = mw + (lane & 15);
    const int kb = (lane >> 4) << 4;     // 0 or 16
    Frag fa0, fa1;
    fa0.q[0] = *(const u32x4*)&Ash[am * ROWS + kb];
    fa0.q[1] = *(const u32x4*)&Ash[am * ROWS + kb + 8];
    fa1.q[0] = *(const u32x4*)&Ash[am * ROWS + 32 + kb];
    fa1.q[1] = *(const u32x4*)&Ash[am * ROWS + 32 + kb + 8];
    const int bn = nw + (lane & 15);
    const int k0 = (lane >> 4) << 4;     // 0 or 16
    Frag fb00, fb01, fb10, fb11;         // [n-half][k-chunk]
    fb00.q[0] = *(const u32x4*)&Bsh[bn * ROWS + k0];
    fb00.q[1] = *(const u32x4*)&Bsh[bn * ROWS + k0 + 8];
    fb01.q[0] = *(const u32x4*)&Bsh[bn * ROWS + 32 + k0];
    fb01.q[1] = *(const u32x4*)&Bsh[bn * ROWS + 32 + k0 + 8];
    fb10.q[0] = *(const u32x4*)&Bsh[(bn + 16) * ROWS + k0];
    fb10.q[1] = *(const u32x4*)&Bsh[(bn + 16) * ROWS + k0 + 8];
    fb11.q[0] = *(const u32x4*)&Bsh[(bn + 16) * ROWS + 32 + k0];
    fb11.q[1] = *(const u32x4*)&Bsh[(bn + 16) * ROWS + 32 + k0 + 8];

    acc0 = wmma_bf16(fa0, fb00, acc0);
    acc0 = wmma_bf16(fa1, fb01, acc0);
    acc1 = wmma_bf16(fa0, fb10, acc1);
    acc1 = wmma_bf16(fa1, fb11, acc1);
    __syncthreads();
  }

  // epilogue: +bias, ReLU, bf16 store channels-last
  const int nl = lane & 15;
  const int mo = (lane >> 4) << 3;       // upper half lanes hold rows M+8
#pragma unroll
  for (int r = 0; r < 8; ++r) {
    const int mg = (mtile << 6) + mw + r + mo;
    {
      const int ng = (ntile << 6) + nw + nl;
      float v = acc0[r] + bc[ng];
      v = v > 0.0f ? v : 0.0f;
      featws[(size_t)mg * 512 + ng] = f2bf(v);
    }
    {
      const int ng = (ntile << 6) + nw + 16 + nl;
      float v = acc1[r] + bc[ng];
      v = v > 0.0f ? v : 0.0f;
      featws[(size_t)mg * 512 + ng] = f2bf(v);
    }
  }
}

// =====================================================================
// Kernel 2: fused 1x1 convs (cls 18ch + box 36ch, padded to N=64) via WMMA
// featws bf16 [32768,512] -> cls_score fp32 ws (NCHW), box_reg fp32 d_out
// K stepped 64 (8 stages), register double-buffered.
// =====================================================================
__global__ __launch_bounds__(256) void k_heads_wmma(
    const unsigned short* __restrict__ featws,
    const float* __restrict__ Wcls, const float* __restrict__ bcls,
    const float* __restrict__ Wbox, const float* __restrict__ bbox,
    float* __restrict__ clsws, float* __restrict__ outbox) {
  __shared__ __attribute__((aligned(16))) unsigned short Ash[64 * ROWS];
  __shared__ __attribute__((aligned(16))) unsigned short Bsh[64 * ROWS];

  const int tid   = threadIdx.x;
  const int mtile = blockIdx.x;          // 0..511
  const int lane = tid & 31, wv = tid >> 5;
  const int mw = (wv & 3) << 4;
  const int nw = (wv >> 2) << 5;

  v8f acc0 = {}; v8f acc1 = {};

  const int mA   = tid >> 2;             // A-stage: 64 rows
  const int coff = (tid & 3) << 3;       // A-stage: 8-bf16 chunk within 32
  const int nB   = tid & 63;
  const int csB  = tid >> 6;

  u32x4 a0, a1;
  float bR[16];

  auto loadStage = [&](int kt) {
    const int c0 = kt << 6;
    const int mg = (mtile << 6) + mA;
    a0 = *(const u32x4*)(featws + (size_t)mg * 512 + c0 + coff);
    a1 = *(const u32x4*)(featws + (size_t)mg * 512 + c0 + 32 + coff);
#pragma unroll
    for (int j = 0; j < 16; ++j) {
      const int ci = (csB << 4) + j;
      float v = 0.0f;
      if (nB < 18)      v = Wcls[nB * 512 + c0 + ci];
      else if (nB < 54) v = Wbox[(nB - 18) * 512 + c0 + ci];
      bR[j] = v;
    }
  };

  loadStage(0);
  for (int kt = 0; kt < 8; ++kt) {
    *(u32x4*)&Ash[mA * ROWS + coff]      = a0;
    *(u32x4*)&Ash[mA * ROWS + 32 + coff] = a1;
#pragma unroll
    for (int j = 0; j < 16; ++j)
      Bsh[nB * ROWS + (csB << 4) + j] = f2bf(bR[j]);
    __syncthreads();

    if (kt + 1 < 8) loadStage(kt + 1);

    const int am = mw + (lane & 15);
    const int kb = (lane >> 4) << 4;
    Frag fa0, fa1;
    fa0.q[0] = *(const u32x4*)&Ash[am * ROWS + kb];
    fa0.q[1] = *(const u32x4*)&Ash[am * ROWS + kb + 8];
    fa1.q[0] = *(const u32x4*)&Ash[am * ROWS + 32 + kb];
    fa1.q[1] = *(const u32x4*)&Ash[am * ROWS + 32 + kb + 8];
    const int bn = nw + (lane & 15);
    const int k0 = (lane >> 4) << 4;
    Frag fb00, fb01, fb10, fb11;
    fb00.q[0] = *(const u32x4*)&Bsh[bn * ROWS + k0];
    fb00.q[1] = *(const u32x4*)&Bsh[bn * ROWS + k0 + 8];
    fb01.q[0] = *(const u32x4*)&Bsh[bn * ROWS + 32 + k0];
    fb01.q[1] = *(const u32x4*)&Bsh[bn * ROWS + 32 + k0 + 8];
    fb10.q[0] = *(const u32x4*)&Bsh[(bn + 16) * ROWS + k0];
    fb10.q[1] = *(const u32x4*)&Bsh[(bn + 16) * ROWS + k0 + 8];
    fb11.q[0] = *(const u32x4*)&Bsh[(bn + 16) * ROWS + 32 + k0];
    fb11.q[1] = *(const u32x4*)&Bsh[(bn + 16) * ROWS + 32 + k0 + 8];

    acc0 = wmma_bf16(fa0, fb00, acc0);
    acc0 = wmma_bf16(fa1, fb01, acc0);
    acc1 = wmma_bf16(fa0, fb10, acc1);
    acc1 = wmma_bf16(fa1, fb11, acc1);
    __syncthreads();
  }

  const int nl = lane & 15;
  const int mo = (lane >> 4) << 3;
#pragma unroll
  for (int r = 0; r < 8; ++r) {
    const int mg  = (mtile << 6) + mw + r + mo;
    const int bb  = mg >> 12;
    const int pix = mg & 4095;
#pragma unroll
    for (int half = 0; half < 2; ++half) {
      const int ng = nw + half * 16 + nl;
      const float a = half ? acc1[r] : acc0[r];
      if (ng < 18) {
        clsws[(((size_t)bb * 18 + ng) << 12) + pix] = a + bcls[ng];
      } else if (ng < 54) {
        const int c2 = ng - 18;
        outbox[(((size_t)bb * 36 + c2) << 12) + pix] = a + bbox[c2];
      }
    }
  }
}

// =====================================================================
// Kernel 3: 2-way softmax over (bg=a, fg=a+9) channel pairs -> cls_prob
// =====================================================================
__global__ __launch_bounds__(256) void k_softmax2(
    const float* __restrict__ clsws, float* __restrict__ outcls) {
  const int i = blockIdx.x * 256 + threadIdx.x;   // over 8*9*4096
  if (i >= 8 * 9 * 4096) return;
  const int b   = i / (9 * 4096);
  const int rem = i - b * 9 * 4096;
  const int a   = rem >> 12;
  const int pix = rem & 4095;
  const float s0 = clsws[(((size_t)b * 18 + a) << 12) + pix];
  const float s1 = clsws[(((size_t)b * 18 + 9 + a) << 12) + pix];
  const float m  = fmaxf(s0, s1);
  const float e0 = expf(s0 - m), e1 = expf(s1 - m);
  const float inv = 1.0f / (e0 + e1);
  outcls[(((size_t)b * 18 + a) << 12) + pix]     = e0 * inv;
  outcls[(((size_t)b * 18 + 9 + a) << 12) + pix] = e1 * inv;
}

// =====================================================================
// Kernel 4: per-image anchor decode + clip + min-size + greedy NMS (300)
// one 1024-thread block per image; scores/boxes in global ws.
// =====================================================================
__global__ __launch_bounds__(1024) void k_proposal_nms(
    const float* __restrict__ outcls, const float* __restrict__ outbox,
    const float* __restrict__ iminfo, float* __restrict__ boxesws,
    float* __restrict__ scoresws, float* __restrict__ rois) {
  __shared__ float vals[1024];
  __shared__ int   idxs[1024];
  __shared__ float sb[5];
  __shared__ int   sbi;

  const int b   = blockIdx.x;
  const int tid = threadIdx.x;
  const float imh = iminfo[b * 3 + 0];
  const float imw = iminfo[b * 3 + 1];
  const float isc = iminfo[b * 3 + 2];
  const float mx = imw - 1.0f, my = imh - 1.0f;
  const float mins = 16.0f * isc;
  const size_t sbase = (size_t)b * 36864;

  // ---- phase 1: decode all 36864 anchors ----
  for (int k = 0; k < 36; ++k) {
    const int i = tid + (k << 10);
    const int a = i % 9, pix = i / 9;
    const int py = pix >> 6, px = pix & 63;
    const int r_ = a / 3, s_ = a - r_ * 3;
    const float ratio = (r_ == 0) ? 0.5f : ((r_ == 1) ? 1.0f : 2.0f);
    const float scl = (float)(8 << s_);
    const float w0 = roundf(sqrtf(256.0f / ratio));
    const float h0 = roundf(w0 * ratio);
    const float w = w0 * scl, h = h0 * scl;
    const float shx = (float)(px << 4), shy = (float)(py << 4);
    const float ax1 = 7.5f - 0.5f * (w - 1.0f) + shx;
    const float ay1 = 7.5f - 0.5f * (h - 1.0f) + shy;
    const float ax2 = 7.5f + 0.5f * (w - 1.0f) + shx;
    const float ay2 = 7.5f + 0.5f * (h - 1.0f) + shy;
    const float wa = ax2 - ax1 + 1.0f, ha = ay2 - ay1 + 1.0f;
    const float xa = ax1 + 0.5f * wa,  ya = ay1 + 0.5f * ha;
    const float d0 = outbox[(((size_t)b * 36 + a * 4 + 0) << 12) + pix];
    const float d1 = outbox[(((size_t)b * 36 + a * 4 + 1) << 12) + pix];
    const float d2 = outbox[(((size_t)b * 36 + a * 4 + 2) << 12) + pix];
    const float d3 = outbox[(((size_t)b * 36 + a * 4 + 3) << 12) + pix];
    const float cx = d0 * wa + xa, cy = d1 * ha + ya;
    const float pw = expf(d2) * wa, ph = expf(d3) * ha;
    float x1 = fminf(fmaxf(cx - 0.5f * pw, 0.0f), mx);
    float y1 = fminf(fmaxf(cy - 0.5f * ph, 0.0f), my);
    float x2 = fminf(fmaxf(cx + 0.5f * pw, 0.0f), mx);
    float y2 = fminf(fmaxf(cy + 0.5f * ph, 0.0f), my);
    const float wb = x2 - x1 + 1.0f, hb = y2 - y1 + 1.0f;
    const float fg = outcls[(((size_t)b * 18 + 9 + a) << 12) + pix];
    const float sc = (wb >= mins && hb >= mins) ? fg : NEGV;
    f32x4 bx = {x1, y1, x2, y2};
    *(f32x4*)(boxesws + ((sbase + i) << 2)) = bx;
    scoresws[sbase + i] = sc;
  }
  __threadfence_block();
  __syncthreads();

  // ---- phase 2: 300 greedy NMS iterations ----
  for (int it = 0; it < 300; ++it) {
    float bv = -3.0e38f; int bi = 1 << 30;
    for (int k = 0; k < 36; ++k) {
      const int i = tid + (k << 10);
      const float s = scoresws[sbase + i];
      if (s > bv) { bv = s; bi = i; }
    }
    vals[tid] = bv; idxs[tid] = bi;
    __syncthreads();
    for (int off = 512; off > 0; off >>= 1) {
      if (tid < off) {
        const float v2 = vals[tid + off]; const int i2 = idxs[tid + off];
        if (v2 > vals[tid] || (v2 == vals[tid] && i2 < idxs[tid])) {
          vals[tid] = v2; idxs[tid] = i2;
        }
      }
      __syncthreads();
    }
    if (tid == 0) {
      const int besti = idxs[0]; const float bestv = vals[0];
      const f32x4 bb = *(const f32x4*)(boxesws + ((sbase + besti) << 2));
      sb[0] = bb[0]; sb[1] = bb[1]; sb[2] = bb[2]; sb[3] = bb[3]; sb[4] = bestv;
      sbi = besti;
      scoresws[sbase + besti] = NEGV;
      float* r = rois + ((size_t)b * 300 + it) * 5;
      const bool ok = bestv > -5.0e9f;
      r[0] = (float)b;
      r[1] = ok ? bb[0] : 0.0f; r[2] = ok ? bb[1] : 0.0f;
      r[3] = ok ? bb[2] : 0.0f; r[4] = ok ? bb[3] : 0.0f;
    }
    __threadfence_block();
    __syncthreads();
    const float b1 = sb[0], b2 = sb[1], b3 = sb[2], b4 = sb[3];
    const int bI = sbi;
    const float barea = (b3 - b1 + 1.0f) * (b4 - b2 + 1.0f);
    for (int k = 0; k < 36; ++k) {
      const int i = tid + (k << 10);
      if (i == bI) continue;
      const float s = scoresws[sbase + i];
      if (s > -5.0e9f) {
        const f32x4 bb = *(const f32x4*)(boxesws + ((sbase + i) << 2));
        const float xx1 = fmaxf(b1, bb[0]);
        const float yy1 = fmaxf(b2, bb[1]);
        const float xx2 = fminf(b3, bb[2]);
        const float yy2 = fminf(b4, bb[3]);
        const float inter = fmaxf(xx2 - xx1 + 1.0f, 0.0f) *
                            fmaxf(yy2 - yy1 + 1.0f, 0.0f);
        const float area = (bb[2] - bb[0] + 1.0f) * (bb[3] - bb[1] + 1.0f);
        const float iou = inter / (barea + area - inter);
        if (iou > 0.7f) scoresws[sbase + i] = NEGV;
      }
    }
    __threadfence_block();
    __syncthreads();
  }
}

// =====================================================================
extern "C" void kernel_launch(void* const* d_in, const int* in_sizes, int n_in,
                              void* d_out, int out_size, void* d_ws, size_t ws_size,
                              hipStream_t stream) {
  (void)in_sizes; (void)n_in; (void)out_size; (void)ws_size;
  const float* features = (const float*)d_in[0];
  // d_in[1] = gt_boxes (unused in eval-mode forward)
  const float* im_info  = (const float*)d_in[2];
  const float* Wc       = (const float*)d_in[3];
  const float* bc       = (const float*)d_in[4];
  const float* Wcls     = (const float*)d_in[5];
  const float* bcls     = (const float*)d_in[6];
  const float* Wbox     = (const float*)d_in[7];
  const float* bbox     = (const float*)d_in[8];

  float* out = (float*)d_out;
  float* out_rois = out;                              // [8,300,5]    = 12000
  float* out_cls  = out + 12000;                      // [8,18,64,64] = 589824
  float* out_box  = out + 12000 + 589824;             // [8,36,64,64] = 1179648

  char* ws = (char*)d_ws;
  unsigned short* featws = (unsigned short*)(ws + 0);            // 33,554,432 B
  float* clsws    = (float*)(ws + 33554432);                     //  2,359,296 B
  float* boxesws  = (float*)(ws + 35913728);                     //  4,718,592 B
  float* scoresws = (float*)(ws + 40632320);                     //  1,179,648 B

  k_conv3x3_wmma<<<dim3(512, 8), 256, 0, stream>>>(features, Wc, bc, featws);
  k_heads_wmma<<<512, 256, 0, stream>>>(featws, Wcls, bcls, Wbox, bbox,
                                        clsws, out_box);
  k_softmax2<<<(8 * 9 * 4096 + 255) / 256, 256, 0, stream>>>(clsws, out_cls);
  k_proposal_nms<<<8, 1024, 0, stream>>>(out_cls, out_box, im_info,
                                         boxesws, scoresws, out_rois);
}
